// MambaVisionMixer3D_39702677684812
// MI455X (gfx1250) — matrix operand order repaired and checked by reading, not compiled, not run on hardware
//
#include <hip/hip_runtime.h>
#include <hip/hip_bf16.h>
#include <math.h>

// ---------------------------------------------------------------------------
// MambaVision 3D mixer for gfx1250 (MI455X).
//  - in_proj / out_proj GEMMs: v_wmma_f32_16x16x32_bf16, 32x32 wave tile,
//    operands pre-converted to bf16 (halves GEMM operand traffic).
//  - selective scan: 2-pass segmented scan (16 segments x 256 steps) to cut
//    the serial exp/log chain 16x and raise wave occupancy 16x.
//  - permute / dwconv+silu / x_proj / dt_proj: f32 VALU, memory-bound.
// ---------------------------------------------------------------------------

#define B_    2
#define L_    4096
#define DM_   512
#define HALF_ 256
#define NST_  16
#define RNK_  32
#define NDIR_ 3
#define E_    (NDIR_ * 2 * HALF_)   // 1536
#define NSEG_ 16
#define SEGL_ (L_ / NSEG_)          // 256

typedef __attribute__((ext_vector_type(16))) __bf16 v16bf;
typedef __attribute__((ext_vector_type(8)))  __bf16 v8bf;
typedef __attribute__((ext_vector_type(8)))  float  v8f;

// ---------------- scan-order permutations (16x16x16 cube) -----------------
__device__ __forceinline__ int fwd_perm(int dir, int l) {
    if (dir == 0) return l;
    int d = l >> 8, h = (l >> 4) & 15, w = l & 15;
    if (dir == 1) return (w << 8) | (d << 4) | h;   // (w,d,h)
    return (h << 8) | (w << 4) | d;                  // (h,w,d)
}
__device__ __forceinline__ int inv_perm(int dir, int ls) {
    if (dir == 0) return ls;
    if (dir == 1) { int w = ls >> 8, d = (ls >> 4) & 15, h = ls & 15;
                    return (d << 8) | (h << 4) | w; }
    int h = ls >> 8, w = (ls >> 4) & 15, d = ls & 15;
    return (d << 8) | (h << 4) | w;
}

// ---------------- bf16 WMMA fragment loads ---------------------------------
// A fragment (16x32, row-major MxK, ld=K):
//  lane: row = lane&15 ; elems 0..7 -> K = k0 + 8*(lane>>4) + e ; 8..15 -> +16
__device__ __forceinline__ v16bf load_a_frag(const __bf16* __restrict__ A,
                                             int ld, int row, int k0, int lane) {
    const __bf16* p = A + (size_t)row * ld + k0 + ((lane >> 4) << 3);
    v8bf lo = *(const v8bf*)(p);
    v8bf hi = *(const v8bf*)(p + 16);
    v16bf f;
#pragma unroll
    for (int e = 0; e < 8; e++) { f[e] = lo[e]; f[8 + e] = hi[e]; }
    return f;
}
// B fragment (32x16 of W^T, W row-major NxK, ld=K):
//  lane: col n = lane&15 (-> W row) ; elems 0..7 -> K = k0 + 8*(lane>>4) + e ; 8..15 -> +16
__device__ __forceinline__ v16bf load_b_frag(const __bf16* __restrict__ W,
                                             int ld, int row, int k0, int lane) {
    const __bf16* p = W + (size_t)row * ld + k0 + ((lane >> 4) << 3);
    v8bf lo = *(const v8bf*)(p);
    v8bf hi = *(const v8bf*)(p + 16);
    v16bf f;
#pragma unroll
    for (int e = 0; e < 8; e++) { f[e] = lo[e]; f[8 + e] = hi[e]; }
    return f;
}

// ---------------- GEMM: C[MxN] = A[MxK] * W[NxK]^T  (bf16 in, f32 accum) ---
// One wave computes a 32(M) x 32(N) tile: 2 A-frags x 2 B-frags, 4 WMMAs/step.
__global__ void gemm_bf16_wmma(const __bf16* __restrict__ A,
                               const __bf16* __restrict__ W,
                               float* __restrict__ C,
                               int M, int N, int K) {
    int lane = threadIdx.x & 31;
    int wid  = blockIdx.x * (blockDim.x >> 5) + (threadIdx.x >> 5);
    int tN   = N >> 5;
    int totW = (M >> 5) * tN;
    if (wid >= totW) return;
    int tm = wid / tN, tn = wid % tN;
    int m0 = tm << 5, n0 = tn << 5;

    int arow0 = m0 + (lane & 15), arow1 = arow0 + 16;
    int brow0 = n0 + (lane & 15), brow1 = brow0 + 16;

    v8f c00 = {}, c01 = {}, c10 = {}, c11 = {};
    for (int k0 = 0; k0 < K; k0 += 32) {
        v16bf a0 = load_a_frag(A, K, arow0, k0, lane);
        v16bf a1 = load_a_frag(A, K, arow1, k0, lane);
        v16bf b0 = load_b_frag(W, K, brow0, k0, lane);
        v16bf b1 = load_b_frag(W, K, brow1, k0, lane);
        c00 = __builtin_amdgcn_wmma_f32_16x16x32_bf16(false, a0, false, b0, (short)0, c00, false, false);
        c01 = __builtin_amdgcn_wmma_f32_16x16x32_bf16(false, a0, false, b1, (short)0, c01, false, false);
        c10 = __builtin_amdgcn_wmma_f32_16x16x32_bf16(false, a1, false, b0, (short)0, c10, false, false);
        c11 = __builtin_amdgcn_wmma_f32_16x16x32_bf16(false, a1, false, b1, (short)0, c11, false, false);
    }
    // C frag: VGPR r -> M = r + 8*(lane>>4), N = lane&15
    int crow = m0 + ((lane >> 4) << 3);
    int ccol = n0 + (lane & 15);
#pragma unroll
    for (int r = 0; r < 8; r++) {
        C[(size_t)(crow + r) * N + ccol]           = c00[r];
        C[(size_t)(crow + r) * N + ccol + 16]      = c01[r];
        C[(size_t)(crow + r + 16) * N + ccol]      = c10[r];
        C[(size_t)(crow + r + 16) * N + ccol + 16] = c11[r];
    }
}

// ---------------- f32 -> bf16 conversion -----------------------------------
__global__ void f32_to_bf16(const float* __restrict__ in,
                            __bf16* __restrict__ out, int n) {
    int idx = blockIdx.x * blockDim.x + threadIdx.x;
    if (idx < n) out[idx] = (__bf16)in[idx];
}

// ---------------- split xz -> x,z in scan domain ---------------------------
__global__ void permute_split(const float* __restrict__ xz,
                              float* __restrict__ xs, float* __restrict__ zs) {
    int idx = blockIdx.x * blockDim.x + threadIdx.x;   // (b,l,i,c) c fastest
    int c  = idx & 255;
    int i  = (idx >> 8) % 3;
    int bl = idx / 768;
    int l  = bl & (L_ - 1);
    int b  = bl >> 12;
    size_t src = ((size_t)(b * L_ + l)) * E_ + i * 512 + c;
    int ls = fwd_perm(i, l);
    size_t dst = ((size_t)((b * NDIR_ + i) * HALF_ + c)) * L_ + ls;
    xs[dst] = xz[src];
    zs[dst] = xz[src + 256];
}

// ---------------- depthwise conv(K=3, same) + SiLU, x branch ---------------
__global__ void conv_silu_x(const float* __restrict__ xs,
                            const float* __restrict__ w,
                            const float* __restrict__ bias,
                            float* __restrict__ xc) {
    int idx = blockIdx.x * blockDim.x + threadIdx.x;   // (b,i,c,l) l fastest
    int l  = idx & (L_ - 1);
    int c  = (idx >> 12) & 255;
    int bi = idx >> 20;
    int i  = bi % 3;
    const float* row = xs + ((size_t)bi * HALF_ + c) * L_;
    const float* wk  = w + (i * HALF_ + c) * 3;
    float v = wk[1] * row[l] + bias[i * HALF_ + c];
    if (l > 0)      v += wk[0] * row[l - 1];
    if (l < L_ - 1) v += wk[2] * row[l + 1];
    v = v / (1.0f + expf(-v));
    xc[idx] = v;
}

// ---------------- z branch: conv+silu, scatter bf16 into y_comb ------------
__global__ void conv_silu_z(const float* __restrict__ zs,
                            const float* __restrict__ w,
                            const float* __restrict__ bias,
                            __bf16* __restrict__ ycomb) {
    int idx = blockIdx.x * blockDim.x + threadIdx.x;
    int l  = idx & (L_ - 1);
    int c  = (idx >> 12) & 255;
    int bi = idx >> 20;
    int i  = bi % 3;
    int b  = bi / 3;
    const float* row = zs + ((size_t)bi * HALF_ + c) * L_;
    const float* wk  = w + (i * HALF_ + c) * 3;
    float v = wk[1] * row[l] + bias[i * HALF_ + c];
    if (l > 0)      v += wk[0] * row[l - 1];
    if (l < L_ - 1) v += wk[2] * row[l + 1];
    v = v / (1.0f + expf(-v));
    int lo = inv_perm(i, l);
    ycomb[((size_t)(b * L_ + lo)) * E_ + i * 512 + 256 + c] = (__bf16)v;
}

// ---------------- x_proj ---------------------------------------------------
__global__ void xproj_kernel(const float* __restrict__ xc,
                             const float* __restrict__ xpw,
                             float* __restrict__ xdbl) {
    int idx = blockIdx.x * blockDim.x + threadIdx.x;   // (b,i,o,l) l fastest
    int l  = idx & (L_ - 1);
    int o  = (idx >> 12) & 63;
    int bi = idx >> 18;
    int i  = bi % 3;
    const float* xrow = xc + (size_t)bi * HALF_ * L_;
    const float* wrow = xpw + (i * 64 + o) * HALF_;
    float s = 0.f;
#pragma unroll 4
    for (int d = 0; d < HALF_; d++)
        s += wrow[d] * xrow[(size_t)d * L_ + l];
    xdbl[((size_t)bi * 64 + o) * L_ + l] = s;
}

// ---------------- dt_proj --------------------------------------------------
__global__ void dtproj_kernel(const float* __restrict__ xdbl,
                              const float* __restrict__ dtw,
                              float* __restrict__ dtf) {
    int idx = blockIdx.x * blockDim.x + threadIdx.x;   // (b,i,d,l) l fastest
    int l  = idx & (L_ - 1);
    int dd = (idx >> 12) & 255;
    int bi = idx >> 20;
    int i  = bi % 3;
    const float* base = xdbl + (size_t)bi * 64 * L_;
    const float* wrow = dtw + (i * HALF_ + dd) * RNK_;
    float s = 0.f;
#pragma unroll
    for (int r = 0; r < RNK_; r++)
        s += wrow[r] * base[(size_t)r * L_ + l];
    dtf[idx] = s;
}

// ---------------- segmented selective scan ---------------------------------
// h_l = a_l*h_{l-1} + bu_l composes per segment as (Aseg = prod a,
// Bseg = segment-local end state). Pass 1 emits per-segment (A,B); pass 2
// folds predecessor segments for its initial h, re-runs the local recurrence
// and scatters y (bf16) into y_comb. Lane map: wave = 2 channels x 16 states.
__device__ __forceinline__ void scan_lane_setup(int blk, int tid,
        int& b, int& i, int& seg, int& d, int& n, int& bi) {
    seg = blk & (NSEG_ - 1);
    int t = blk >> 4;
    int grp = t & 15;
    int t2 = t >> 4;
    i = t2 % 3; b = t2 / 3;
    int lane = tid & 31, wave = tid >> 5;
    d = grp * 16 + wave * 2 + (lane >> 4);
    n = lane & 15;
    bi = b * NDIR_ + i;
}

__global__ void scan_pass1(const float* __restrict__ xc,
                           const float* __restrict__ dtf,
                           const float* __restrict__ xdbl,
                           const float* __restrict__ A_log,
                           const float* __restrict__ dt_bias,
                           float* __restrict__ Aseg,
                           float* __restrict__ Bseg) {
    int b, i, seg, d, n, bi;
    scan_lane_setup(blockIdx.x, threadIdx.x, b, i, seg, d, n, bi);
    float Av   = -expf(A_log[(i * HALF_ + d) * NST_ + n]);
    float bias = dt_bias[i * HALF_ + d];
    const float* up  = xc   + ((size_t)bi * HALF_ + d) * L_;
    const float* dtp = dtf  + ((size_t)bi * HALF_ + d) * L_;
    const float* Bpp = xdbl + ((size_t)bi * 64 + 32 + n) * L_;
    float Ap = 1.f, Bp = 0.f;
    int l0 = seg * SEGL_;
    for (int l = l0; l < l0 + SEGL_; l++) {
        float u   = up[l];
        float dtv = dtp[l] + bias;
        float delta = (dtv > 20.f) ? dtv : log1pf(expf(dtv));
        float a  = expf(delta * Av);
        float bu = delta * Bpp[l] * u;
        Ap *= a;
        Bp = a * Bp + bu;
    }
    size_t sidx = (((size_t)bi * HALF_ + d) * NST_ + n) * NSEG_ + seg;
    Aseg[sidx] = Ap;
    Bseg[sidx] = Bp;
}

__global__ void scan_pass2(const float* __restrict__ xc,
                           const float* __restrict__ dtf,
                           const float* __restrict__ xdbl,
                           const float* __restrict__ A_log,
                           const float* __restrict__ D_param,
                           const float* __restrict__ dt_bias,
                           const float* __restrict__ Aseg,
                           const float* __restrict__ Bseg,
                           __bf16* __restrict__ ycomb) {
    int b, i, seg, d, n, bi;
    scan_lane_setup(blockIdx.x, threadIdx.x, b, i, seg, d, n, bi);
    float Av   = -expf(A_log[(i * HALF_ + d) * NST_ + n]);
    float Dv   = D_param[i * HALF_ + d];
    float bias = dt_bias[i * HALF_ + d];
    const float* up  = xc   + ((size_t)bi * HALF_ + d) * L_;
    const float* dtp = dtf  + ((size_t)bi * HALF_ + d) * L_;
    const float* Bpp = xdbl + ((size_t)bi * 64 + 32 + n) * L_;
    const float* Cpp = xdbl + ((size_t)bi * 64 + 48 + n) * L_;
    // fold predecessor segments to get this segment's initial state
    size_t sbase = (((size_t)bi * HALF_ + d) * NST_ + n) * NSEG_;
    float h = 0.f;
    for (int k = 0; k < seg; k++)
        h = Aseg[sbase + k] * h + Bseg[sbase + k];
    int l0 = seg * SEGL_;
    for (int l = l0; l < l0 + SEGL_; l++) {
        float u   = up[l];
        float dtv = dtp[l] + bias;
        float delta = (dtv > 20.f) ? dtv : log1pf(expf(dtv));
        h = expf(delta * Av) * h + delta * Bpp[l] * u;
        float p = h * Cpp[l];
        p += __shfl_xor(p, 1, 32);
        p += __shfl_xor(p, 2, 32);
        p += __shfl_xor(p, 4, 32);
        p += __shfl_xor(p, 8, 32);
        if (n == 0) {
            int lo = inv_perm(i, l);
            ycomb[((size_t)(b * L_ + lo)) * E_ + i * 512 + d] = (__bf16)(p + Dv * u);
        }
    }
}

// ---------------------------------------------------------------------------
extern "C" void kernel_launch(void* const* d_in, const int* in_sizes, int n_in,
                              void* d_out, int out_size, void* d_ws, size_t ws_size,
                              hipStream_t stream) {
    const float* hidden    = (const float*)d_in[0];   // (B,L,DM)
    const float* in_proj_w = (const float*)d_in[1];   // (1536,512)
    const float* x_proj_w  = (const float*)d_in[2];   // (3,64,256)
    const float* dt_proj_w = (const float*)d_in[3];   // (3,256,32)
    const float* dt_proj_b = (const float*)d_in[4];   // (3,256)
    const float* A_log     = (const float*)d_in[5];   // (3,256,16)
    const float* D_param   = (const float*)d_in[6];   // (3,256)
    const float* conv_x_w  = (const float*)d_in[7];   // (3,256,1,3)
    const float* conv_x_b  = (const float*)d_in[8];   // (3,256)
    const float* conv_z_w  = (const float*)d_in[9];
    const float* conv_z_b  = (const float*)d_in[10];
    const float* out_proj_w= (const float*)d_in[11];  // (512,1536)
    float* out = (float*)d_out;                       // (B,L,DM)

    // ---- workspace layout (bytes) ----
    const size_t N_XZ  = (size_t)B_ * L_ * E_;              // 12,582,912
    const size_t N_DIR = (size_t)B_ * NDIR_ * HALF_ * L_;   //  6,291,456
    const size_t N_SEG = (size_t)B_ * NDIR_ * HALF_ * NST_ * NSEG_; // 393,216
    char* base = (char*)d_ws;
    float*  xz       = (float*)base;                        // [N_XZ] f32
    __bf16* ycomb    = (__bf16*)base;                       // alias: xz dead after permute
    base += N_XZ * sizeof(float);
    float*  xs       = (float*)base;  base += N_DIR * sizeof(float);
    float*  zs       = (float*)base;  base += N_DIR * sizeof(float);
    float*  xdbl     = zs;                                   // alias: zs dead after conv_z
    float*  xc       = (float*)base;  base += N_DIR * sizeof(float);
    float*  dtf      = (float*)base;  base += N_DIR * sizeof(float);
    float*  Aseg     = (float*)base;  base += N_SEG * sizeof(float);
    float*  Bseg     = (float*)base;  base += N_SEG * sizeof(float);
    __bf16* hiddenb  = (__bf16*)base; base += (size_t)B_ * L_ * DM_ * sizeof(__bf16);
    __bf16* inwb     = (__bf16*)base; base += (size_t)E_ * DM_ * sizeof(__bf16);
    __bf16* outwb    = (__bf16*)base; base += (size_t)DM_ * E_ * sizeof(__bf16);

    const int T = 256;

    // 0) pre-convert GEMM operands to bf16
    {
        int n = B_ * L_ * DM_;
        f32_to_bf16<<<(n + T - 1) / T, T, 0, stream>>>(hidden, hiddenb, n);
        n = E_ * DM_;
        f32_to_bf16<<<(n + T - 1) / T, T, 0, stream>>>(in_proj_w, inwb, n);
        n = DM_ * E_;
        f32_to_bf16<<<(n + T - 1) / T, T, 0, stream>>>(out_proj_w, outwb, n);
    }
    // 1) in_proj GEMM: xz[8192,1536] = hidden @ in_proj_w^T
    {
        int M = B_ * L_, N = E_, K = DM_;
        int waves = (M / 32) * (N / 32);
        gemm_bf16_wmma<<<(waves + 7) / 8, 256, 0, stream>>>(hiddenb, inwb, xz, M, N, K);
    }
    // 2) split + forward scan permutation
    permute_split<<<(int)(B_ * L_ * NDIR_ * HALF_ / T), T, 0, stream>>>(xz, xs, zs);
    // 3) depthwise conv + silu
    conv_silu_x<<<(int)(N_DIR / T), T, 0, stream>>>(xs, conv_x_w, conv_x_b, xc);
    conv_silu_z<<<(int)(N_DIR / T), T, 0, stream>>>(zs, conv_z_w, conv_z_b, ycomb);
    // 4) x_proj -> x_dbl (dtl | B | C)
    xproj_kernel<<<(int)((size_t)B_ * NDIR_ * 64 * L_ / T), T, 0, stream>>>(xc, x_proj_w, xdbl);
    // 5) dt_proj
    dtproj_kernel<<<(int)(N_DIR / T), T, 0, stream>>>(xdbl, dt_proj_w, dtf);
    // 6) segmented selective scan (two passes), scatter y (bf16) into y_comb
    {
        int blocks = B_ * NDIR_ * 16 * NSEG_;   // 1536
        scan_pass1<<<blocks, 256, 0, stream>>>(xc, dtf, xdbl, A_log, dt_proj_b, Aseg, Bseg);
        scan_pass2<<<blocks, 256, 0, stream>>>(xc, dtf, xdbl, A_log, D_param, dt_proj_b,
                                               Aseg, Bseg, ycomb);
    }
    // 7) out_proj GEMM: out[8192,512] = y_comb @ out_proj_w^T
    {
        int M = B_ * L_, N = DM_, K = E_;
        int waves = (M / 32) * (N / 32);
        gemm_bf16_wmma<<<(waves + 7) / 8, 256, 0, stream>>>(ycomb, outwb, out, M, N, K);
    }
}
